// HeterogeneousSupervisedTopicModel_18150531793584
// MI455X (gfx1250) — compile-verified, matrix operand order checked by reading, not run on hardware
//
#include <hip/hip_runtime.h>
#include <hip/hip_bf16.h>

// ---------------- problem constants ----------------
#define VV 50000   // vocab
#define VP 50176   // vocab padded: 16 K-splits * 98 iters * 32
#define HH 300     // hidden
#define HP 304     // hidden padded
#define KK 50      // topics
#define KP 64      // topics padded
#define BB 512     // batch
#define BN_EPS 1e-5f
#define KS1 16     // K-splits for GEMM1
#define KS3 16     // K-splits for het GEMM

typedef __bf16 bf16_t;
typedef bf16_t v16bf __attribute__((ext_vector_type(16)));
typedef bf16_t v8bf  __attribute__((ext_vector_type(8)));
typedef float  v8f   __attribute__((ext_vector_type(8)));

// ---------------- workspace layout (all partial buffers fully written -> no zeroing) ----
// float region
#define OFF_H1P    0                              // h1part [KS1][BB*HP]
#define OFF_HETP   (OFF_H1P + KS1*BB*HP)          // hetpart[KS3][BB*KP]
#define OFF_H1     (OFF_HETP + KS3*BB*KP)
#define OFF_H2     (OFF_H1 + BB*HH)
#define OFF_TH     (OFF_H2 + BB*HH)
#define OFF_RM     (OFF_TH + BB*KK)
#define OFF_RS     (OFF_RM + BB)
#define OFF_BD     (OFF_RS + BB)
#define OFF_RECONP (OFF_BD + BB)                  // [2048] per-wave recon partials
#define OFF_KLDP   (OFF_RECONP + 2048)            // [BB]
#define OFF_MSEP   (OFF_KLDP + BB)                // [BB]
#define OFF_L1BP   (OFF_MSEP + BB)                // [512]
#define OFF_L1GP   (OFF_L1BP + 512)               // [512]
#define OFF_SSBP   (OFF_L1GP + 512)               // [512]
#define N_FLOATS   (OFF_SSBP + 512)
// bf16 region (element offsets from bf16 base = ws + N_FLOATS)
#define OFF16_NB   0                       // nb16   [512][VP]
#define OFF16_W1   (OFF16_NB + BB*VP)      // W116   [300][VP]
#define OFF16_BETA (OFF16_W1 + HH*VP)      // betaPad[VV][64]
#define OFF16_SBT  (OFF16_BETA + VV*KP)    // sbT    [50][VP]
#define OFF16_THP  (OFF16_SBT + KK*VP)     // thetaPad[512][64]

// ---------------- WMMA helpers ----------------
__device__ __forceinline__ v8f zero8(){
  v8f c;
#pragma unroll
  for(int i=0;i<8;++i) c[i]=0.f;
  return c;
}
// 16-bit fragment, K contiguous bf16 (16B-aligned):
// lanes 0-15 hold K=[0..7]+[16..23], lanes 16-31 hold K=[8..15]+[24..31].
__device__ __forceinline__ v16bf frag_bf16(const bf16_t* p, int half){
  const bf16_t* q = p + half*8;
  v8bf lo = *(const v8bf*)q;
  v8bf hi = *(const v8bf*)(q + 16);
  return __builtin_shufflevector(lo, hi, 0,1,2,3,4,5,6,7,8,9,10,11,12,13,14,15);
}
__device__ __forceinline__ v8f wmma_bf16(v16bf a, v16bf b, v8f c){
  return __builtin_amdgcn_wmma_f32_16x16x32_bf16(false, a, false, b, (short)0, c, false, false);
}

// ---------------- prep kernels (fp32 -> padded bf16, one-shot) ----------------
__global__ void k_prep_rowpad(const float* __restrict__ src, bf16_t* __restrict__ dst,
                              int rows, int cols){
  size_t i = (size_t)blockIdx.x*blockDim.x + threadIdx.x;
  if(i >= (size_t)rows*VP) return;
  int row = (int)(i / VP), k = (int)(i % VP);
  dst[i] = (k < cols) ? (bf16_t)src[(size_t)row*cols + k] : (bf16_t)0.f;
}
__global__ void k_prep_beta(const float* __restrict__ lb, bf16_t* __restrict__ betaPad){
  size_t i = (size_t)blockIdx.x*blockDim.x + threadIdx.x;
  if(i >= (size_t)VV*KP) return;
  int v = (int)(i >> 6), k = (int)(i & 63);
  betaPad[i] = (k < KK) ? (bf16_t)lb[(size_t)v*KK + k] : (bf16_t)0.f;
}
__global__ void k_prep_sbt(const float* __restrict__ lb, const float* __restrict__ gm,
                           bf16_t* __restrict__ sbT){
  int k = blockIdx.x*blockDim.x + threadIdx.x;
  if(k >= VP) return;
  if(k < VV){
    const float* l = lb + (size_t)k*KK;
    const float* g = gm + (size_t)k*KK;
    for(int c=0;c<KK;++c) sbT[(size_t)c*VP + k] = (bf16_t)(l[c]*g[c]);
  } else {
    for(int c=0;c<KK;++c) sbT[(size_t)c*VP + k] = (bf16_t)0.f;
  }
}

// ---------------- GEMM1: h1part[ks] = nb @ W1.T (per-K-split, deterministic) ----------
// Wave tile 32Mx64N (8 wmma/step). 8 waves/WG share the 64-col B chunk staged to LDS
// via global_load_async_to_lds_b128 (double-buffered, ASYNCcnt + split barrier).
#define LDSB 40   // padded LDS row stride (bf16): 80B -> bank-conflict-free, 16B-aligned
__global__ __launch_bounds__(256) void k_gemm1(const bf16_t* __restrict__ nb16,
                                               const bf16_t* __restrict__ W116,
                                               float* __restrict__ h1part){
  __shared__ bf16_t bufB[2][64*LDSB];
  const int tid = threadIdx.x;
  const int wid = tid >> 5, lane = tid & 31, half = lane >> 4, r = lane & 15;
  const int wg  = blockIdx.x;            // 160 = 2 wgM * 5 ng * 16 ks
  const int ks  = wg & 15;
  const int ng  = (wg >> 4) % 5;
  const int wgM = wg / 80;               // 0..1
  const int mg  = wgM*8 + wid;           // 0..15 -> rows mg*32..mg*32+31
  const int kbase = ks * (VP/KS1);       // 3136
  const int iters = (VP/KS1)/32;         // 98

  // B staging: one async b128 (8 bf16) per thread per K-step
  const int scol = tid >> 2;             // 0..63
  const int skof = (tid & 3) * 8;        // 0,8,16,24
  int sgcol = ng*64 + scol; if(sgcol >= HH) sgcol = HH-1;   // clamp (store-guarded)
  const bf16_t* sgptr = W116 + (size_t)sgcol*VP + kbase + skof;
  {
    unsigned lo = (unsigned)(size_t)(void*)&bufB[0][scol*LDSB + skof];
    asm volatile("global_load_async_to_lds_b128 %0, %1, off" :: "v"(lo), "v"(sgptr) : "memory");
  }

  const bf16_t* arow0 = nb16 + (size_t)(mg*32 +      r)*VP + kbase;
  const bf16_t* arow1 = nb16 + (size_t)(mg*32 + 16 + r)*VP + kbase;

  v8f c[2][4];
#pragma unroll
  for(int mt=0; mt<2; ++mt)
#pragma unroll
    for(int nt=0; nt<4; ++nt) c[mt][nt] = zero8();

  for(int it=0; it<iters; ++it){
    asm volatile("s_wait_asynccnt 0x0" ::: "memory");   // current stage landed
    __syncthreads();
    if(it+1 < iters){
      unsigned lo = (unsigned)(size_t)(void*)&bufB[(it+1)&1][scol*LDSB + skof];
      const bf16_t* gp = sgptr + (size_t)(it+1)*32;
      asm volatile("global_load_async_to_lds_b128 %0, %1, off" :: "v"(lo), "v"(gp) : "memory");
    }
    __builtin_prefetch(arow0 + (size_t)it*32 + 256, 0, 1);
    __builtin_prefetch(arow1 + (size_t)it*32 + 256, 0, 1);
    v16bf a0 = frag_bf16(arow0 + (size_t)it*32, half);
    v16bf a1 = frag_bf16(arow1 + (size_t)it*32, half);
    const bf16_t* bb = &bufB[it&1][0];
#pragma unroll
    for(int nt=0; nt<4; ++nt){
      v16bf b = frag_bf16(bb + (nt*16 + r)*LDSB, half);
      c[0][nt] = wmma_bf16(a0, b, c[0][nt]);
      c[1][nt] = wmma_bf16(a1, b, c[1][nt]);
    }
  }
  float* dst = h1part + (size_t)ks*(BB*HP);
#pragma unroll
  for(int nt=0; nt<4; ++nt){
    int col = ng*64 + nt*16 + r;
    if(col < HH){
#pragma unroll
      for(int mt=0; mt<2; ++mt)
#pragma unroll
        for(int i=0;i<8;++i){
          int m = mg*32 + mt*16 + i + half*8;     // C layout: vgpr i -> M=i / M=i+8
          dst[(size_t)m*HP + col] = c[mt][nt][i]; // unique writer -> plain store
        }
    }
  }
}

// ---------------- encoder epilogues / heads (deterministic fixed-order sums) ---------
__global__ void k_enc1_epi(const float* h1part, const float* b1,
                           const float* g, const float* bb, const float* m, const float* v,
                           float* h1){
  int i = blockIdx.x*blockDim.x + threadIdx.x;
  if(i >= BB*HH) return;
  int b = i / HH, j = i % HH;
  float x = b1[j];
  for(int ks=0; ks<KS1; ++ks) x += h1part[(size_t)ks*(BB*HP) + (size_t)b*HP + j];
  x = fmaxf(x, 0.f);
  h1[i] = (x - m[j]) * rsqrtf(v[j] + BN_EPS) * g[j] + bb[j];
}
__global__ void k_enc2(const float* h1, const float* W2, const float* b2,
                       const float* g, const float* bb, const float* m, const float* v,
                       float* h2){
  int i = blockIdx.x*blockDim.x + threadIdx.x;
  if(i >= BB*HH) return;
  int b = i / HH, j = i % HH;
  const float* hr = h1 + (size_t)b*HH;
  const float* wr = W2 + (size_t)j*HH;
  float acc = b2[j];
  for(int t=0;t<HH;++t) acc = fmaf(hr[t], wr[t], acc);
  acc = fmaxf(acc, 0.f);
  h2[i] = (acc - m[j]) * rsqrtf(v[j] + BN_EPS) * g[j] + bb[j];
}
__global__ __launch_bounds__(64) void k_head(const float* h2,
                                             const float* muW, const float* mub,
                                             const float* lsW, const float* lsb,
                                             float* theta, bf16_t* thetaPad, float* kldp){
  int b = blockIdx.x, t = threadIdx.x;
  __shared__ float smu[KK];
  __shared__ float sk[64];
  __shared__ float sred[2];
  const float* hr = h2 + (size_t)b*HH;
  float kld = 0.f;
  if(t < KK){
    const float* wm = muW + (size_t)t*HH;
    const float* wl = lsW + (size_t)t*HH;
    float mu = mub[t], ls = lsb[t];
    for(int i=0;i<HH;++i){ float h = hr[i]; mu = fmaf(h, wm[i], mu); ls = fmaf(h, wl[i], ls); }
    smu[t] = mu;
    kld = 1.f + ls - mu*mu - __expf(ls);
  }
  sk[t] = kld;
  __syncthreads();
  if(t == 0){
    float s = 0.f;
    for(int i=0;i<64;++i) s += sk[i];
    kldp[b] = s;                                  // per-row partial, fixed order
    float mx = -__builtin_inff();
    for(int i=0;i<KK;++i) mx = fmaxf(mx, smu[i]);
    float den = 0.f;
    for(int i=0;i<KK;++i) den += __expf(smu[i]-mx);
    sred[0] = mx; sred[1] = den;
  }
  __syncthreads();
  float th = (t < KK) ? (__expf(smu[t]-sred[0]) / sred[1]) : 0.f;
  if(t < KK) theta[(size_t)b*KK + t] = th;
  thetaPad[(size_t)b*KP + t] = (bf16_t)th;
}

// ---------------- decode pass 1: online per-row (max, sumexp) over V ----------------
__global__ __launch_bounds__(512) void k_dec_pass1(const bf16_t* __restrict__ thetaPad,
                                                   const bf16_t* __restrict__ betaPad,
                                                   const float* __restrict__ base,
                                                   float* row_max, float* row_sum){
  int tm = blockIdx.x;
  int wid = threadIdx.x >> 5;
  int lane = threadIdx.x & 31, half = lane >> 4, r = lane & 15;
  __shared__ float sm[16][16], ss[16][16];

  const bf16_t* arow = thetaPad + (size_t)(tm*16 + r)*KP;
  v16bf a0 = frag_bf16(arow,      half);
  v16bf a1 = frag_bf16(arow + 32, half);

  float mr[8], sr[8];
#pragma unroll
  for(int i=0;i<8;++i){ mr[i] = -__builtin_inff(); sr[i] = 0.f; }

  for(int vt = wid; vt < VV/16; vt += 16){
    int v = vt*16 + r;
    const bf16_t* bcol = betaPad + (size_t)v*KP;
    v16bf b0 = frag_bf16(bcol,      half);
    v16bf b1 = frag_bf16(bcol + 32, half);
    v8f c = zero8();
    c = wmma_bf16(a0, b0, c);
    c = wmma_bf16(a1, b1, c);
    float br = base[v];
#pragma unroll
    for(int i=0;i<8;++i){
      float l = c[i] + br;
      float nm = fmaxf(mr[i], l);
      sr[i] = sr[i]*__expf(mr[i]-nm) + __expf(l-nm);
      mr[i] = nm;
    }
  }
#pragma unroll
  for(int i=0;i<8;++i){
    for(int d=1; d<16; d<<=1){
      float om = __shfl_xor(mr[i], d, 32);
      float os = __shfl_xor(sr[i], d, 32);
      float nm = fmaxf(mr[i], om);
      sr[i] = sr[i]*__expf(mr[i]-nm) + os*__expf(om-nm);
      mr[i] = nm;
    }
  }
  if(r == 0){
#pragma unroll
    for(int i=0;i<8;++i){ sm[wid][half*8+i] = mr[i]; ss[wid][half*8+i] = sr[i]; }
  }
  __syncthreads();
  if(threadIdx.x < 16){
    int lr = threadIdx.x;
    float m = -__builtin_inff(), s = 0.f;
    for(int w=0; w<16; ++w){
      float om = sm[w][lr], os = ss[w][lr];
      float nm = fmaxf(m, om);
      s = s*__expf(m-nm) + os*__expf(om-nm);
      m = nm;
    }
    row_max[tm*16+lr] = m;
    row_sum[tm*16+lr] = s;
  }
}

// ---------------- decode pass 2: log(softmax+1e-6)*bows, per-wave partials ----------
__global__ __launch_bounds__(256) void k_dec_pass2(const bf16_t* __restrict__ thetaPad,
                                                   const bf16_t* __restrict__ betaPad,
                                                   const float* __restrict__ base,
                                                   const float* __restrict__ bows,
                                                   const float* row_max, const float* row_sum,
                                                   float* reconp){
  int gw = blockIdx.x*8 + (threadIdx.x>>5);   // 2048 waves = 32 M-tiles x 64 V-splits
  int lane = threadIdx.x & 31, half = lane>>4, r = lane&15;
  int tm = gw >> 6;
  int part = gw & 63;
  int rowbase = tm*16;
  const bf16_t* arow = thetaPad + (size_t)(rowbase + r)*KP;
  v16bf a0 = frag_bf16(arow,      half);
  v16bf a1 = frag_bf16(arow + 32, half);
  float rm[8], si[8];
#pragma unroll
  for(int i=0;i<8;++i){
    int m = rowbase + i + half*8;
    rm[i] = row_max[m];
    si[i] = 1.f / row_sum[m];
  }
  float acc = 0.f;
  for(int vt = part; vt < VV/16; vt += 64){
    int v = vt*16 + r;
    const bf16_t* bcol = betaPad + (size_t)v*KP;
    v16bf b0 = frag_bf16(bcol,      half);
    v16bf b1 = frag_bf16(bcol + 32, half);
    v8f c = zero8();
    c = wmma_bf16(a0, b0, c);
    c = wmma_bf16(a1, b1, c);
    float br = base[v];
#pragma unroll
    for(int i=0;i<8;++i){
      float l = c[i] + br;
      float p = __expf(l - rm[i]) * si[i];
      float w = bows[(size_t)(rowbase + i + half*8)*VV + v];
      acc = fmaf(__logf(p + 1e-6f), w, acc);
    }
  }
  for(int d=1; d<32; d<<=1) acc += __shfl_xor(acc, d, 32);
  if(lane == 0) reconp[gw] = acc;               // per-wave partial, fixed order later
}

// ---------------- het GEMM: hetpart[ks] = nb @ (beta*gamma) ----------------
__global__ __launch_bounds__(256) void k_het(const bf16_t* __restrict__ nb16,
                                             const bf16_t* __restrict__ sbT,
                                             float* __restrict__ hetpart){
  int gw = blockIdx.x*8 + (threadIdx.x>>5);   // 512 waves = 32 tm x 16 ks
  int lane = threadIdx.x & 31, half = lane>>4, r = lane&15;
  int ks = gw & 15;
  int tm = gw >> 4;
  const int kbase = ks * (VP/KS3);   // 3136
  const int iters = (VP/KS3)/32;     // 98
  const bf16_t* arow = nb16 + (size_t)(tm*16 + r)*VP + kbase;
  const bf16_t* bptr[4];
#pragma unroll
  for(int nt=0; nt<4; ++nt){
    int col = nt*16 + r; if(col >= KK) col = KK-1;       // clamp (store-guarded)
    bptr[nt] = sbT + (size_t)col*VP + kbase;
  }
  v8f c[4];
#pragma unroll
  for(int nt=0; nt<4; ++nt) c[nt] = zero8();
  for(int it=0; it<iters; ++it){
    __builtin_prefetch(arow + (size_t)it*32 + 256, 0, 1);
    v16bf a = frag_bf16(arow + (size_t)it*32, half);
#pragma unroll
    for(int nt=0; nt<4; ++nt){
      v16bf b = frag_bf16(bptr[nt] + (size_t)it*32, half);
      c[nt] = wmma_bf16(a, b, c[nt]);
    }
  }
  float* dst = hetpart + (size_t)ks*(BB*KP);
#pragma unroll
  for(int nt=0; nt<4; ++nt){
    int col = nt*16 + r;
    if(col < KK){
#pragma unroll
      for(int i=0;i<8;++i){
        int m = tm*16 + i + half*8;
        dst[(size_t)m*KP + col] = c[nt][i];     // unique writer -> plain store
      }
    }
  }
}

// ---------------- small tail kernels (all deterministic) ----------------
__global__ __launch_bounds__(256) void k_bowdot(const float* nb, const float* bw, float* bd){
  int b = blockIdx.x;
  const float* row = nb + (size_t)b*VV;
  float acc = 0.f;
  for(int v = threadIdx.x; v < VV; v += 256) acc = fmaf(row[v], bw[v], acc);
  __shared__ float s[256];
  s[threadIdx.x] = acc; __syncthreads();
  for(int d=128; d>0; d>>=1){ if(threadIdx.x<d) s[threadIdx.x]+=s[threadIdx.x+d]; __syncthreads(); }
  if(threadIdx.x == 0) bd[b] = s[0];
}
__device__ __forceinline__ float block_reduce256(float v, float* s){
  s[threadIdx.x] = v; __syncthreads();
  for(int d=128; d>0; d>>=1){ if(threadIdx.x<d) s[threadIdx.x]+=s[threadIdx.x+d]; __syncthreads(); }
  float r = s[0]; __syncthreads();
  return r;
}
__global__ __launch_bounds__(256) void k_reg(const float* gammas, const float* base,
                                             const float* bw,
                                             float* l1gp, float* l1bp, float* ssbp){
  __shared__ float s[256];
  const size_t stride = 512*256;
  size_t gid = (size_t)blockIdx.x*256 + threadIdx.x;
  float vg = 0.f;
  for(size_t i = gid; i < (size_t)VV*KK; i += stride) vg += fabsf(gammas[i]);
  float t = block_reduce256(vg, s);
  if(threadIdx.x == 0) l1gp[blockIdx.x] = t;
  float vb = 0.f;
  for(size_t i = gid; i < VV; i += stride) vb += fabsf(base[i]);
  t = block_reduce256(vb, s);
  if(threadIdx.x == 0) l1bp[blockIdx.x] = t;
  float vw = 0.f;
  for(size_t i = gid; i < VV; i += stride) vw += bw[i]*bw[i];
  t = block_reduce256(vw, s);
  if(threadIdx.x == 0) ssbp[blockIdx.x] = t;
}
__global__ __launch_bounds__(512) void k_pred(const float* hetpart, const float* theta,
                                              const float* bd, const float* bow_b,
                                              const float* topic_w, const float* topic_b,
                                              const float* labels, float* msep){
  int b = threadIdx.x;
  float e = bd[b] + bow_b[0] + topic_b[0];
  const float* th = theta + (size_t)b*KK;
  for(int k=0;k<KK;++k){
    float hsum = 0.f;
    for(int ks=0; ks<KS3; ++ks) hsum += hetpart[(size_t)ks*(BB*KP) + (size_t)b*KP + k];
    e = fmaf(hsum + topic_w[k], th[k], e);
  }
  float d = e - labels[b];
  msep[b] = d*d;
}
// single-block deterministic final reduction + scalar assembly
__global__ __launch_bounds__(256) void k_final(const float* reconp, const float* kldp,
                                               const float* msep, const float* l1bp,
                                               const float* l1gp, const float* ssbp,
                                               float* out){
  __shared__ float s[256];
  int t = threadIdx.x;
  float a;
  a = 0.f; for(int i=t; i<2048; i+=256) a += reconp[i];
  float recon = block_reduce256(a, s);
  a = 0.f; for(int i=t; i<BB;   i+=256) a += kldp[i];
  float kld   = block_reduce256(a, s);
  a = 0.f; for(int i=t; i<BB;   i+=256) a += msep[i];
  float mse   = block_reduce256(a, s);
  a = 0.f; for(int i=t; i<512;  i+=256) a += l1bp[i];
  float l1b   = block_reduce256(a, s);
  a = 0.f; for(int i=t; i<512;  i+=256) a += l1gp[i];
  float l1g   = block_reduce256(a, s);
  a = 0.f; for(int i=t; i<512;  i+=256) a += ssbp[i];
  float ssb   = block_reduce256(a, s);
  if(t == 0){
    out[0] = -recon / (float)BB;
    out[1] = 5e-4f*l1b + mse/(float)BB + 5e-6f*l1g + 5e-4f*sqrtf(ssb);
    out[2] = -0.5f * kld / (float)BB;
  }
}

// ---------------- host launch ----------------
extern "C" void kernel_launch(void* const* d_in, const int* in_sizes, int n_in,
                              void* d_out, int out_size, void* d_ws, size_t ws_size,
                              hipStream_t stream){
  const float* bows   = (const float*)d_in[0];
  const float* nb     = (const float*)d_in[1];
  const float* labels = (const float*)d_in[2];
  const float* W1     = (const float*)d_in[3];
  const float* b1     = (const float*)d_in[4];
  const float* W2     = (const float*)d_in[5];
  const float* b2     = (const float*)d_in[6];
  const float* bn1_g  = (const float*)d_in[7];
  const float* bn1_b  = (const float*)d_in[8];
  const float* bn1_m  = (const float*)d_in[9];
  const float* bn1_v  = (const float*)d_in[10];
  const float* bn2_g  = (const float*)d_in[11];
  const float* bn2_b  = (const float*)d_in[12];
  const float* bn2_m  = (const float*)d_in[13];
  const float* bn2_v  = (const float*)d_in[14];
  const float* muW    = (const float*)d_in[15];
  const float* mub    = (const float*)d_in[16];
  const float* lsW    = (const float*)d_in[17];
  const float* lsb    = (const float*)d_in[18];
  const float* lb     = (const float*)d_in[19];
  const float* gm     = (const float*)d_in[20];
  const float* base   = (const float*)d_in[21];
  const float* bw     = (const float*)d_in[22];
  const float* bowb   = (const float*)d_in[23];
  const float* tw     = (const float*)d_in[24];
  const float* tb     = (const float*)d_in[25];

  float* ws      = (float*)d_ws;
  float* h1part  = ws + OFF_H1P;
  float* hetpart = ws + OFF_HETP;
  float* h1      = ws + OFF_H1;
  float* h2      = ws + OFF_H2;
  float* theta   = ws + OFF_TH;
  float* rmax    = ws + OFF_RM;
  float* rsum    = ws + OFF_RS;
  float* bd      = ws + OFF_BD;
  float* reconp  = ws + OFF_RECONP;
  float* kldp    = ws + OFF_KLDP;
  float* msep    = ws + OFF_MSEP;
  float* l1bp    = ws + OFF_L1BP;
  float* l1gp    = ws + OFF_L1GP;
  float* ssbp    = ws + OFF_SSBP;
  bf16_t* b16    = (bf16_t*)(ws + N_FLOATS);
  bf16_t* nb16   = b16 + OFF16_NB;
  bf16_t* W116   = b16 + OFF16_W1;
  bf16_t* betaPad= b16 + OFF16_BETA;
  bf16_t* sbT    = b16 + OFF16_SBT;
  bf16_t* thetaPad = b16 + OFF16_THP;
  float* out     = (float*)d_out;

  // 0) one-shot bf16 conversions (padded K); no zeroing needed (partials fully written)
  k_prep_rowpad<<<(BB*VP+255)/256, 256, 0, stream>>>(nb, nb16, BB, VV);
  k_prep_rowpad<<<(HH*VP+255)/256, 256, 0, stream>>>(W1, W116, HH, VV);
  k_prep_beta<<<(VV*KP+255)/256, 256, 0, stream>>>(lb, betaPad);
  k_prep_sbt<<<(VP+255)/256, 256, 0, stream>>>(lb, gm, sbT);
  // 1) encoder GEMM1 (bf16 WMMA, async-LDS B staging, deterministic K-split partials)
  k_gemm1<<<160, 256, 0, stream>>>(nb16, W116, h1part);
  k_enc1_epi<<<(BB*HH+255)/256, 256, 0, stream>>>(h1part, b1, bn1_g, bn1_b, bn1_m, bn1_v, h1);
  // 2) encoder stage-2 + heads
  k_enc2<<<(BB*HH+255)/256, 256, 0, stream>>>(h1, W2, b2, bn2_g, bn2_b, bn2_m, bn2_v, h2);
  k_head<<<BB, 64, 0, stream>>>(h2, muW, mub, lsW, lsb, theta, thetaPad, kldp);
  // 3) decode: streamed softmax over V (WMMA twice, nothing [B,V] materialized)
  k_dec_pass1<<<32, 512, 0, stream>>>(thetaPad, betaPad, base, rmax, rsum);
  k_dec_pass2<<<256, 256, 0, stream>>>(thetaPad, betaPad, base, bows, rmax, rsum, reconp);
  // 4) het GEMM + bow dot + regularizers + prediction MSE
  k_het<<<64, 256, 0, stream>>>(nb16, sbT, hetpart);
  k_bowdot<<<BB, 256, 0, stream>>>(nb, bw, bd);
  k_reg<<<512, 256, 0, stream>>>(gm, base, bw, l1gp, l1bp, ssbp);
  k_pred<<<1, 512, 0, stream>>>(hetpart, theta, bd, bowb, tw, tb, labels, msep);
  // 5) deterministic final reduction
  k_final<<<1, 256, 0, stream>>>(reconp, kldp, msep, l1bp, l1gp, ssbp, out);
}
